// ToyMultiHeadAttention_43181601194157
// MI455X (gfx1250) — matrix operand order repaired and checked by reading, not compiled, not run on hardware
//
#include <hip/hip_runtime.h>
#include <hip/hip_bf16.h>

typedef __bf16 bf16;
typedef __attribute__((ext_vector_type(16))) __bf16 v16bf;
typedef __attribute__((ext_vector_type(8)))  float  v8f;

#define Bn 4
#define Tn 2048
#define Cn 1024
#define Hn 16
#define Dn 64

__device__ __forceinline__ v8f wmma_bf16f32(v16bf a, v16bf b, v8f c) {
  return __builtin_amdgcn_wmma_f32_16x16x32_bf16(false, a, false, b, (short)0, c,
                                                 false, false);
}

union Frag {
  v16bf v;
  unsigned int u[8];
};

// ---- CDNA5 async global->LDS copy (ASYNCcnt-tracked, bypasses VGPR file) ----
__device__ __forceinline__ void async_ld_b128(void* lds, const void* g) {
  asm volatile("global_load_async_to_lds_b128 %0, %1, off"
               :: "v"((unsigned)(unsigned long long)lds),
                  "v"((unsigned long long)g)
               : "memory");
}
__device__ __forceinline__ void async_wait0() {
  asm volatile("s_wait_asynccnt 0x0" ::: "memory");
}

// ---- row reductions over 16-lane halves via v_permlane16_b32 (VALU, no LDS) --
__device__ __forceinline__ float pl16(float x, unsigned lo, unsigned hi) {
  unsigned xi = __float_as_uint(x);
  unsigned r = __builtin_amdgcn_permlane16(xi, xi, lo, hi, false, false);
  return __uint_as_float(r);
}
__device__ __forceinline__ float rowmax16(float x) {
  x = fmaxf(x, pl16(x, 0x67452301u, 0xEFCDAB89u));  // lane ^ 1
  x = fmaxf(x, pl16(x, 0x54761032u, 0xDCFE98BAu));  // lane ^ 2
  x = fmaxf(x, pl16(x, 0x32107654u, 0xBA98FEDCu));  // lane ^ 4
  x = fmaxf(x, pl16(x, 0xFEDCBA98u, 0x76543210u));  // lane ^ 8
  return x;
}
__device__ __forceinline__ float rowsum16(float x) {
  x += pl16(x, 0x67452301u, 0xEFCDAB89u);
  x += pl16(x, 0x54761032u, 0xDCFE98BAu);
  x += pl16(x, 0x32107654u, 0xBA98FEDCu);
  x += pl16(x, 0xFEDCBA98u, 0x76543210u);
  return x;
}

// ---------------------------------------------------------------- fp32 -> bf16
__global__ void cvt_f32_bf16(const float* __restrict__ x, bf16* __restrict__ y,
                             int n4) {
  int i = blockIdx.x * blockDim.x + threadIdx.x;
  if (i < n4) {
    const float4 f = ((const float4*)x)[i];
    union { bf16 h[4]; uint2 u; } pk;
    pk.h[0] = (bf16)f.x; pk.h[1] = (bf16)f.y;
    pk.h[2] = (bf16)f.z; pk.h[3] = (bf16)f.w;
    *(uint2*)&y[(size_t)i * 4] = pk.u;
  }
}

// ------------------------------------------ GEMM: out = (A @ W^T + bias)*scale
// OM=0: bf16 [B,H,T,D];  OM=1: bf16 [B,H,D,T] (V transposed);  OM=2: f32 [M,C].
// Block: 128x128 tile, 8 waves (2x4), 64x32 per wave. Async double-buffered.
template <int OM>
__global__ __launch_bounds__(256) void gemm_xwT(const bf16* __restrict__ A,
                                                const bf16* __restrict__ W,
                                                const float* __restrict__ bias,
                                                void* __restrict__ outp,
                                                float scale) {
  __shared__ bf16 As[2][128][40];
  __shared__ bf16 Bs[2][128][40];

  const int tid  = threadIdx.x;
  const int m0   = blockIdx.x * 128;
  const int n0   = blockIdx.y * 128;
  const int lane = tid & 31;
  const int wv   = tid >> 5;
  const int wr   = wv >> 2;      // wave row 0..1 (64 rows)
  const int wc   = wv & 3;       // wave col 0..3 (32 cols)
  const int n16  = lane & 15;
  const int hf   = lane >> 4;

  auto issue = [&](int buf, int k0) {
#pragma unroll
    for (int it = 0; it < 2; ++it) {
      int idx = tid + it * 256;
      int row = idx >> 2, seg = idx & 3;
      async_ld_b128(&As[buf][row][seg * 8],
                    &A[(size_t)(m0 + row) * Cn + k0 + seg * 8]);
      async_ld_b128(&Bs[buf][row][seg * 8],
                    &W[(size_t)(n0 + row) * Cn + k0 + seg * 8]);
    }
  };

  v8f acc[4][2] = {};
  int cur = 0;
  issue(0, 0);
  for (int k0 = 0; k0 < Cn; k0 += 32) {
    async_wait0();
    __syncthreads();
    if (k0 + 32 < Cn) issue(cur ^ 1, k0 + 32);

    Frag a[4], b[2];
#pragma unroll
    for (int i = 0; i < 4; ++i)
#pragma unroll
      for (int v = 0; v < 8; ++v) {
        int kk = 2 * (v & 3) + 16 * (v >> 2) + 8 * hf;  // A 16x32 bf16 layout
        a[i].u[v] = *(const unsigned int*)&As[cur][wr * 64 + i * 16 + n16][kk];
      }
#pragma unroll
    for (int j = 0; j < 2; ++j)
#pragma unroll
      for (int v = 0; v < 8; ++v) {
        int kk = 2 * (v & 3) + 16 * (v >> 2) + 8 * hf;  // B 32x16 bf16 layout
        b[j].u[v] = *(const unsigned int*)&Bs[cur][wc * 32 + j * 16 + n16][kk];
      }
#pragma unroll
    for (int i = 0; i < 4; ++i)
#pragma unroll
      for (int j = 0; j < 2; ++j)
        acc[i][j] = wmma_bf16f32(a[i].v, b[j].v, acc[i][j]);
    cur ^= 1;
  }

#pragma unroll
  for (int i = 0; i < 4; ++i)
#pragma unroll
    for (int j = 0; j < 2; ++j) {
      int ng = n0 + wc * 32 + j * 16 + n16;
      float bv = bias[ng];
#pragma unroll
      for (int r = 0; r < 8; ++r) {
        int mg = m0 + wr * 64 + i * 16 + r + 8 * hf;   // C layout: M = r + 8*hf
        float val = (acc[i][j][r] + bv) * scale;
        if (OM == 2) {
          ((float*)outp)[(size_t)mg * Cn + ng] = val;
        } else {
          int t = mg & (Tn - 1), bb = mg >> 11;
          int h = ng >> 6, d = ng & 63;
          if (OM == 0)
            ((bf16*)outp)[(((size_t)bb * Hn + h) * Tn + t) * Dn + d] = (bf16)val;
          else
            ((bf16*)outp)[(((size_t)bb * Hn + h) * Dn + d) * Tn + t] = (bf16)val;
        }
      }
    }
}

// ----------------------------------------------- flash attention (causal)
// Block: 64 query rows of one (b,h). 4 waves x 16 rows. Key tiles of 64.
// Q pre-scaled by D^-0.5. K in [B,H,T,D]; V pre-transposed in [B,H,D,T].
// Only the final key tile touches the diagonal -> mask-free main loop.
__global__ __launch_bounds__(128) void attn_fa(const bf16* __restrict__ Qh,
                                               const bf16* __restrict__ Kh,
                                               const bf16* __restrict__ Vt,
                                               bf16* __restrict__ Out) {
  __shared__ bf16 Ks[2][64][72];   // [key][d]
  __shared__ bf16 Vs[2][64][72];   // [d][key]  (already transposed in memory)
  __shared__ bf16 Ps[4][16][72];   // per-wave P tile [m][key]

  const int tid  = threadIdx.x;
  const int w    = tid >> 5;
  const int lane = tid & 31;
  const int n16  = lane & 15;
  const int hf   = lane >> 4;
  const int q0   = blockIdx.x * 64;
  const int bh   = blockIdx.y;
  const bf16* Qp  = Qh + (size_t)bh * Tn * Dn;
  const bf16* Kp  = Kh + (size_t)bh * Tn * Dn;
  const bf16* Vtp = Vt + (size_t)bh * Dn * Tn;
  const int qr0 = q0 + w * 16;

  // Q fragments (16 rows x 64 d -> two 16x32 A-frags), gathered from global
  Frag qa[2];
#pragma unroll
  for (int p = 0; p < 2; ++p)
#pragma unroll
    for (int v = 0; v < 8; ++v) {
      int kk = 2 * (v & 3) + 16 * (v >> 2) + 8 * hf + p * 32;
      qa[p].u[v] = *(const unsigned int*)&Qp[(size_t)(qr0 + n16) * Dn + kk];
    }

  auto issue = [&](int buf, int k0) {
#pragma unroll
    for (int it = 0; it < 4; ++it) {
      int idx = tid + it * 128;
      int row = idx >> 3, seg = idx & 7;
      async_ld_b128(&Ks[buf][row][seg * 8],
                    &Kp[(size_t)(k0 + row) * Dn + seg * 8]);
      async_ld_b128(&Vs[buf][row][seg * 8],
                    &Vtp[(size_t)row * Tn + k0 + seg * 8]);
    }
  };

  float mrow[8], lrow[8];
#pragma unroll
  for (int r = 0; r < 8; ++r) { mrow[r] = -3.0e38f; lrow[r] = 0.f; }
  v8f o[4] = {};

  auto do_tile = [&](int k0, int cur, bool masked) {
    // S = Q K^T : four 16x16 key tiles, each chaining two K=32 WMMAs over d
    v8f s[4] = {};
#pragma unroll
    for (int j = 0; j < 4; ++j)
#pragma unroll
      for (int p = 0; p < 2; ++p) {
        Frag kb;
#pragma unroll
        for (int v = 0; v < 8; ++v)
          kb.u[v] = *(const unsigned int*)&Ks[cur][j * 16 + n16]
                                             [2 * (v & 3) + 16 * (v >> 2) + 8 * hf + p * 32];
        s[j] = wmma_bf16f32(qa[p].v, kb.v, s[j]);
      }

    // online softmax; each row lives in one 16-lane half -> permlane16 xor net
#pragma unroll
    for (int r = 0; r < 8; ++r) {
      const int qrow = qr0 + r + 8 * hf;
      float x[4];
#pragma unroll
      for (int j = 0; j < 4; ++j) {
        x[j] = s[j][r];
        if (masked && (k0 + j * 16 + n16 > qrow)) x[j] = -3.0e38f;
      }
      float tmax  = rowmax16(fmaxf(fmaxf(x[0], x[1]), fmaxf(x[2], x[3])));
      float mnew  = fmaxf(mrow[r], tmax);
      float alpha = __expf(mrow[r] - mnew);
      float p0 = __expf(x[0] - mnew), p1 = __expf(x[1] - mnew);
      float p2 = __expf(x[2] - mnew), p3 = __expf(x[3] - mnew);
      float ps = rowsum16((p0 + p1) + (p2 + p3));
      lrow[r] = lrow[r] * alpha + ps;
      mrow[r] = mnew;
#pragma unroll
      for (int dt = 0; dt < 4; ++dt) o[dt][r] *= alpha;
      Ps[w][r + 8 * hf][n16]      = (bf16)p0;   // C-layout -> LDS
      Ps[w][r + 8 * hf][16 + n16] = (bf16)p1;
      Ps[w][r + 8 * hf][32 + n16] = (bf16)p2;
      Ps[w][r + 8 * hf][48 + n16] = (bf16)p3;
    }
    __builtin_amdgcn_wave_barrier();

    // O += P V : P re-read in A-frag layout; V^T rows give contiguous B-frags
    Frag pa[2];
#pragma unroll
    for (int pp = 0; pp < 2; ++pp)
#pragma unroll
      for (int v = 0; v < 8; ++v) {
        int kk = 2 * (v & 3) + 16 * (v >> 2) + 8 * hf + 32 * pp;
        pa[pp].u[v] = *(const unsigned int*)&Ps[w][n16][kk];
      }
#pragma unroll
    for (int dt = 0; dt < 4; ++dt)
#pragma unroll
      for (int pp = 0; pp < 2; ++pp) {
        Frag vb;
#pragma unroll
        for (int v = 0; v < 8; ++v)
          vb.u[v] = *(const unsigned int*)&Vs[cur][dt * 16 + n16]
                                             [2 * (v & 3) + 16 * (v >> 2) + 8 * hf + 32 * pp];
        o[dt] = wmma_bf16f32(pa[pp].v, vb.v, o[dt]);
      }
  };

  const int ktiles = (q0 >> 6) + 1;   // 64-wide key tiles, causal
  int cur = 0;
  issue(0, 0);
  for (int kt = 0; kt < ktiles - 1; ++kt) {
    async_wait0();
    __syncthreads();
    issue(cur ^ 1, (kt + 1) * 64);
    do_tile(kt * 64, cur, /*masked=*/false);   // strictly below diagonal
    cur ^= 1;
  }
  async_wait0();
  __syncthreads();
  do_tile((ktiles - 1) * 64, cur, /*masked=*/true);  // diagonal tile

  // normalize + write bf16 in [B,T,C] (merged heads) for the output GEMM
  const int b = bh >> 4, h = bh & 15;
#pragma unroll
  for (int r = 0; r < 8; ++r) {
    float inv = 1.0f / lrow[r];
    int qrow = qr0 + r + 8 * hf;
#pragma unroll
    for (int dt = 0; dt < 4; ++dt) {
      int d = dt * 16 + n16;
      Out[((size_t)b * Tn + qrow) * Cn + h * Dn + d] = (bf16)(o[dt][r] * inv);
    }
  }
}

// ---------------------------------------------------------------- launcher
extern "C" void kernel_launch(void* const* d_in, const int* in_sizes, int n_in,
                              void* d_out, int out_size, void* d_ws,
                              size_t ws_size, hipStream_t stream) {
  (void)in_sizes; (void)n_in; (void)out_size; (void)ws_size;
  const float* q  = (const float*)d_in[0];
  const float* k  = (const float*)d_in[1];
  const float* v  = (const float*)d_in[2];
  const float* Wq = (const float*)d_in[3];
  const float* bq = (const float*)d_in[4];
  const float* Wk = (const float*)d_in[5];
  const float* bk = (const float*)d_in[6];
  const float* Wv = (const float*)d_in[7];
  const float* bv = (const float*)d_in[8];
  const float* Wo = (const float*)d_in[9];
  const float* bo = (const float*)d_in[10];

  const size_t NE = (size_t)Bn * Tn * Cn;  // 8,388,608 elems (16 MB bf16)
  const size_t WE = (size_t)Cn * Cn;       // 1,048,576 elems
  bf16* bufA = (bf16*)d_ws;                // ping-pong buffers, reused
  bf16* bufB = bufA + NE;
  bf16* bufC = bufB + NE;
  bf16* bufD = bufC + NE;
  bf16* w16  = bufD + NE;                  // 4 bf16 weight matrices

  auto cvt = [&](const float* src, bf16* dst, size_t n) {
    int n4 = (int)(n / 4);
    cvt_f32_bf16<<<dim3((n4 + 255) / 256), dim3(256), 0, stream>>>(src, dst, n4);
  };
  cvt(q, bufA, NE);
  cvt(k, bufB, NE);
  cvt(v, bufC, NE);
  cvt(Wq, w16 + 0 * WE, WE);
  cvt(Wk, w16 + 1 * WE, WE);
  cvt(Wv, w16 + 2 * WE, WE);
  cvt(Wo, w16 + 3 * WE, WE);

  dim3 gg(64, 8), gb(256);
  // Q projection pre-scaled by D^-0.5 = 0.125 (folds softmax scale)
  gemm_xwT<0><<<gg, gb, 0, stream>>>(bufA, w16 + 0 * WE, bq, bufD, 0.125f);
  gemm_xwT<0><<<gg, gb, 0, stream>>>(bufB, w16 + 1 * WE, bk, bufA, 1.0f);
  gemm_xwT<1><<<gg, gb, 0, stream>>>(bufC, w16 + 2 * WE, bv, bufB, 1.0f);
  attn_fa<<<dim3(Tn / 64, Bn * Hn), dim3(128), 0, stream>>>(bufD, bufA, bufB,
                                                            bufC);
  gemm_xwT<2><<<gg, gb, 0, stream>>>(bufC, w16 + 3 * WE, bo, d_out, 1.0f);
}
